// DiagnosticDeltaModel_73813307949242
// MI455X (gfx1250) — matrix operand order repaired and checked by reading, not compiled, not run on hardware
//
#include <hip/hip_runtime.h>
#include <hip/hip_bf16.h>

// ---------------------------------------------------------------------------
// Sizes (match reference)
// ---------------------------------------------------------------------------
#define Hdim 128
#define Vdim 32000
#define Bdim 32
#define Ldim 4096
#define TWO_H 256

typedef __attribute__((ext_vector_type(16))) __bf16 v16bf;
typedef __attribute__((ext_vector_type(8)))  __bf16 v8bf;
typedef __attribute__((ext_vector_type(8)))  float  v8f;
typedef __attribute__((ext_vector_type(4)))  float  v4f;
typedef __attribute__((ext_vector_type(4)))  unsigned int u32x4;
typedef __attribute__((ext_vector_type(8)))  int i32x8;
typedef __attribute__((ext_vector_type(4)))  int i32x4;

#if defined(__has_builtin)
#if __has_builtin(__builtin_amdgcn_tensor_load_to_lds) && \
    __has_builtin(__builtin_amdgcn_s_wait_tensorcnt)
#define USE_TDM 1
#endif
#endif

// ---------------------------------------------------------------------------
// WMMA helper (bf16 in, f32 accumulate). wave32 layouts as loaded in this file (contiguous k):
//  A 16x32: lanes 0-15 row M=lane, a[0..15]=K0..15;
//           lanes 16-31 row M=lane-16, a[0..15]=K16..31.
//  B 32x16: lanes 0-15 col N=lane, b[e]=B[K=e][N]; lanes16-31 b[e]=B[K=16+e][N].
//  C 16x16 f32: v8f, lanes0-15 (M=r,N=lane), lanes16-31 (M=8+r,N=lane-16).
// ---------------------------------------------------------------------------
__device__ __forceinline__ v8f wmma_bf(const v16bf& a, const v16bf& b, const v8f& c) {
  return __builtin_amdgcn_wmma_f32_16x16x32_bf16(false, a, false, b, (short)0, c,
                                                 false, false);
}

__device__ __forceinline__ v16bf load_a_bf16v(const __bf16* src, int stride, int k0,
                                              int lid) {
  int row = lid & 15;
  int kh  = (lid < 16) ? 0 : 16;
  const __bf16* p = src + row * stride + k0 + kh;
  v8bf lo = *(const v8bf*)(p);
  v8bf hi = *(const v8bf*)(p + 8);
  return __builtin_shufflevector(lo, hi, 0, 1, 2, 3, 4, 5, 6, 7,
                                 8, 9, 10, 11, 12, 13, 14, 15);
}

__device__ __forceinline__ v16bf load_a_f32v(const float* src, int stride, int k0,
                                             int lid, float scale) {
  int row = lid & 15;
  int kh  = (lid < 16) ? 0 : 16;
  const float* p = src + row * stride + k0 + kh;
  v4f x0 = *(const v4f*)(p);
  v4f x1 = *(const v4f*)(p + 4);
  v4f x2 = *(const v4f*)(p + 8);
  v4f x3 = *(const v4f*)(p + 12);
  v16bf a;
#pragma unroll
  for (int i = 0; i < 4; ++i) {
    a[i]      = (__bf16)(scale * x0[i]);
    a[4 + i]  = (__bf16)(scale * x1[i]);
    a[8 + i]  = (__bf16)(scale * x2[i]);
    a[12 + i] = (__bf16)(scale * x3[i]);
  }
  return a;
}

__device__ __forceinline__ v16bf load_b_frag(const __bf16* frag, int ks, int NT,
                                             int nt, int lid) {
  return ((const v16bf*)frag)[(ks * NT + nt) * 32 + lid];
}

__device__ __forceinline__ v16bf load_bT_f32v(const float* src, int stride, int k0,
                                              int nb, int lid) {
  int n  = nb + (lid & 15);
  int ko = (lid < 16) ? 0 : 16;
  const float* p = src + n * stride + k0 + ko;
  v4f x0 = *(const v4f*)(p);
  v4f x1 = *(const v4f*)(p + 4);
  v4f x2 = *(const v4f*)(p + 8);
  v4f x3 = *(const v4f*)(p + 12);
  v16bf b;
#pragma unroll
  for (int i = 0; i < 4; ++i) {
    b[i]      = (__bf16)x0[i];
    b[4 + i]  = (__bf16)x1[i];
    b[8 + i]  = (__bf16)x2[i];
    b[12 + i] = (__bf16)x3[i];
  }
  return b;
}

__device__ __forceinline__ v8f load_c_f32(const float* src, int stride, int nb,
                                          int lid) {
  int col = nb + (lid & 15);
  int rb  = (lid < 16) ? 0 : 8;
  v8f c;
#pragma unroll
  for (int r = 0; r < 8; ++r) c[r] = src[(rb + r) * stride + col];
  return c;
}

__device__ __forceinline__ void store_c_f32(float* dst, int stride, int nb, int lid,
                                            const v8f& c) {
  int col = nb + (lid & 15);
  int rb  = (lid < 16) ? 0 : 8;
#pragma unroll
  for (int r = 0; r < 8; ++r) dst[(rb + r) * stride + col] = c[r];
}

// ---------------------------------------------------------------------------
// Kernel 0: convert Wout (128 x 32000) f32 -> bf16 once per call
// ---------------------------------------------------------------------------
__global__ void k_cvt(const float* __restrict__ src, __bf16* __restrict__ dst, int n) {
  int i = blockIdx.x * blockDim.x + threadIdx.x;
  if (i < n) dst[i] = (__bf16)src[i];
}

// ---------------------------------------------------------------------------
// Kernel 1: token pipeline. 4 waves/WG, 16 tokens/wave, grid = B*L/64.
// ---------------------------------------------------------------------------
__global__ __launch_bounds__(128) void k_tokens(
    const int* __restrict__ seq, const float* __restrict__ embed,
    const float* __restrict__ W1, const float* __restrict__ b1,
    const float* __restrict__ W2, const float* __restrict__ b2,
    const float* __restrict__ ln_g, const float* __restrict__ ln_b,
    const float* __restrict__ Wkp, float* __restrict__ ksall) {
  extern __shared__ char smem[];
  __bf16* W1s = (__bf16*)smem;                  // 128*256 swizzled
  __bf16* W2s = W1s + Hdim * TWO_H;             // 256*128 swizzled
  __bf16* Wks = W2s + TWO_H * Hdim;             // 128*128 swizzled
  float*  b1s = (float*)(Wks + Hdim * Hdim);    // 256
  float*  b2s = b1s + TWO_H;                    // 128
  float*  lgs = b2s + Hdim;                     // 128
  float*  lbs = lgs + Hdim;                     // 128
  float*  hbase   = lbs + Hdim;                 // 4 * 16*132 f32
  __bf16* actbase = (__bf16*)(hbase + 4 * 16 * 132);  // 4 * 16*264 bf16
  __bf16* hnbase  = actbase + 4 * 16 * 264;           // 4 * 16*136 bf16

  int tid = threadIdx.x;
  for (int i = tid; i < Hdim * TWO_H; i += 128) {    // W1: K=128, N=256
    int k = i >> 8, n = i & 255;
    int lane = (n & 15) + (((k >> 4) & 1) << 4);
    W1s[(((k >> 5) * 16 + (n >> 4)) * 32 + lane) * 16 + (k & 15)] = (__bf16)W1[i];
  }
  for (int i = tid; i < TWO_H * Hdim; i += 128) {    // W2: K=256, N=128
    int k = i >> 7, n = i & 127;
    int lane = (n & 15) + (((k >> 4) & 1) << 4);
    W2s[(((k >> 5) * 8 + (n >> 4)) * 32 + lane) * 16 + (k & 15)] = (__bf16)W2[i];
  }
  for (int i = tid; i < Hdim * Hdim; i += 128) {     // Wkp: K=128, N=128
    int k = i >> 7, n = i & 127;
    int lane = (n & 15) + (((k >> 4) & 1) << 4);
    Wks[(((k >> 5) * 8 + (n >> 4)) * 32 + lane) * 16 + (k & 15)] = (__bf16)Wkp[i];
  }
  for (int i = tid; i < TWO_H; i += 128) b1s[i] = b1[i];
  if (tid < Hdim) { b2s[tid] = b2[tid]; lgs[tid] = ln_g[tid]; lbs[tid] = ln_b[tid]; }
  __syncthreads();

  int wid = tid >> 5, lid = tid & 31;
  float*  hbuf = hbase + wid * (16 * 132);
  __bf16* act  = actbase + wid * (16 * 264);
  __bf16* hn   = hnbase + wid * (16 * 136);
  long base = (long)blockIdx.x * 64 + wid * 16;

  for (int r = 0; r < 16; ++r) {
    int tokidx = seq[base + r];
    v4f v = ((const v4f*)(embed + (long)tokidx * Hdim))[lid];
    *(v4f*)(hbuf + r * 132 + lid * 4) = v;
  }

  // ---- layer 1: act = relu(h @ W1 + b1); A-frags hoisted (reused x16) ----
  {
    v16bf a1[4];
#pragma unroll
    for (int ks = 0; ks < 4; ++ks) a1[ks] = load_a_f32v(hbuf, 132, ks * 32, lid, 1.0f);
    for (int nt = 0; nt < 16; ++nt) {
      v8f c = {};
#pragma unroll
      for (int ks = 0; ks < 4; ++ks)
        c = wmma_bf(a1[ks], load_b_frag(W1s, ks, 16, nt, lid), c);
      int col = nt * 16 + (lid & 15);
      float bias = b1s[col];
      int rb = (lid < 16) ? 0 : 8;
#pragma unroll
      for (int r = 0; r < 8; ++r)
        act[(rb + r) * 264 + col] = (__bf16)fmaxf(c[r] + bias, 0.f);
    }
  }

  // ---- layer 2 + residual: hbuf = h + act @ W2 + b2; A-frags hoisted ----
  {
    v16bf a2[8];
#pragma unroll
    for (int ks = 0; ks < 8; ++ks) a2[ks] = load_a_bf16v(act, 264, ks * 32, lid);
    for (int nt = 0; nt < 8; ++nt) {
      v8f c = {};
#pragma unroll
      for (int ks = 0; ks < 8; ++ks)
        c = wmma_bf(a2[ks], load_b_frag(W2s, ks, 8, nt, lid), c);
      int col = nt * 16 + (lid & 15);
      float bias = b2s[col];
      int rb = (lid < 16) ? 0 : 8;
#pragma unroll
      for (int r = 0; r < 8; ++r) {
        int idx = (rb + r) * 132 + col;
        hbuf[idx] = c[r] + bias + hbuf[idx];
      }
    }
  }

  // ---- layernorm per row (2 lanes per row), write bf16 hn ----
  {
    int row = lid >> 1, j0 = (lid & 1) * 64;
    const float* hp = hbuf + row * 132 + j0;
    float s = 0.f, ss = 0.f;
#pragma unroll 8
    for (int j = 0; j < 64; ++j) { float x = hp[j]; s += x; ss += x * x; }
    s  += __shfl_xor(s, 1);
    ss += __shfl_xor(ss, 1);
    float mu  = s * (1.f / 128.f);
    float var = ss * (1.f / 128.f) - mu * mu;
    float rstd = rsqrtf(fmaxf(var, 0.f) + 1e-5f);
    __bf16* hq = hn + row * 136 + j0;
#pragma unroll 8
    for (int j = 0; j < 64; ++j) {
      float x = hp[j];
      hq[j] = (__bf16)((x - mu) * rstd * lgs[j0 + j] + lbs[j0 + j]);
    }
  }

  // ---- k projection: ks = hn @ Wkp; A-frags hoisted; vectorized store ----
  {
    v16bf a3[4];
#pragma unroll
    for (int ks = 0; ks < 4; ++ks) a3[ks] = load_a_bf16v(hn, 136, ks * 32, lid);
    for (int nt = 0; nt < 8; ++nt) {
      v8f c = {};
#pragma unroll
      for (int ks = 0; ks < 4; ++ks)
        c = wmma_bf(a3[ks], load_b_frag(Wks, ks, 8, nt, lid), c);
      store_c_f32(hbuf, 132, nt * 16, lid, c);
    }
  }
  for (int i = lid; i < 16 * Hdim / 4; i += 32) {
    int row = i >> 5, c4 = (i & 31) * 4;
    v4f v = *(const v4f*)(hbuf + row * 132 + c4);
    *(v4f*)(ksall + (base + row) * Hdim + c4) = v;
  }
}

// ---------------------------------------------------------------------------
// k_scan LDS layout (byte offsets; dynamic LDS starts at offset 0)
// ---------------------------------------------------------------------------
#define OFF_M    0u                       // 128*132 f32 = 67584 B
#define OFF_KC0  67584u                   // 16*132 f32  =  8448 B
#define OFF_KC1  76032u                   // 16*132 f32
#define OFF_KN   84480u                   // 16*132 f32
#define OFF_WB   92928u                   // 16*132 f32
#define OFF_AM   101376u                  // 16*16 f32   =  1024 B
#define OFF_KNBF 102400u                  // 16*136 bf16 =  4352 B
#define OFF_KNT  106752u                  // 128*24 bf16 =  6144 B
#define OFF_UT   112896u                  // 128*24 bf16 =  6144 B
#define SCAN_LDS 119040u

#ifdef USE_TDM
// Issue a TDM load of a 16x128 f32 tile -> LDS with 4-dword row padding
// (reproduces the 132-float LDS row stride). D# per ISA 08_async_tensor §8.
// This toolchain's builtin takes 6 args: (g0, g1, g2, g3, g4, cpol).
__device__ __forceinline__ void tdm_load_chunk(const float* gsrc, unsigned lds_off) {
  unsigned long long ga = (unsigned long long)(__UINTPTR_TYPE__)gsrc;
  u32x4 g0;
  g0[0] = 1u;                                          // count=1 (valid D#)
  g0[1] = lds_off;                                     // lds_addr (bytes)
  g0[2] = (unsigned)(ga & 0xffffffffu);                // global_addr[31:0]
  g0[3] = (unsigned)((ga >> 32) & 0x01ffffffu)         // global_addr[56:32]
        | (2u << 30);                                  // type=2 (image)
  i32x8 g1;
  g1[0] = (2 << 16)      // data_size = 4B
        | (1 << 20)      // pad_enable
        | (6 << 22)      // pad_interval: 128 dwords
        | (3 << 25);     // pad_amount: 4 dwords
  g1[1] = (128 << 16);   // tensor_dim0 = 128
  g1[2] = (16 << 16);    // tensor_dim1 = 16
  g1[3] = (128 << 16);   // tile_dim0 = 128
  g1[4] = 16;            // tile_dim1 = 16, tile_dim2 = 0
  g1[5] = 128;           // tensor_dim0_stride = 128
  g1[6] = 0;
  g1[7] = 0;
  i32x4 z4 = {};
  i32x8 z8 = {};
  __builtin_amdgcn_tensor_load_to_lds(g0, g1, z4, z4, z8, 0);
}
#endif

// ---------------------------------------------------------------------------
// Kernel 2: chunked delta-rule scan (chunk C=16). One WG (8 waves) per batch.
// K-chunks streamed into double-buffered LDS via the Tensor Data Mover,
// pipelined one chunk ahead. M (f32, stride 132) resident in LDS throughout.
// ---------------------------------------------------------------------------
__global__ __launch_bounds__(256) void k_scan(const float* __restrict__ ksall,
                                              float* __restrict__ readv) {
  extern __shared__ char smem[];
  float*  M     = (float*)(smem + OFF_M);
  float*  Kn    = (float*)(smem + OFF_KN);
  float*  Wb    = (float*)(smem + OFF_WB);
  float*  Am    = (float*)(smem + OFF_AM);
  __bf16* Knbf  = (__bf16*)(smem + OFF_KNBF);
  __bf16* KnTbf = (__bf16*)(smem + OFF_KNT);
  __bf16* UTbf  = (__bf16*)(smem + OFF_UT);

  int tid = threadIdx.x, wid = tid >> 5, lid = tid & 31;
  int b = blockIdx.x;
  const float* ksb = ksall + (long)b * Ldim * Hdim;

  for (int i = tid; i < 128 * 132; i += 256) M[i] = 0.f;

  const int NCH = Ldim / 16;  // 256; chunks 0..254 full, chunk 255 has 15 rows

#ifdef USE_TDM
  if (wid == 0) tdm_load_chunk(ksb, OFF_KC0);   // preload chunk 0
#endif
  __syncthreads();

  for (int c = 0; c < NCH; ++c) {
    int t0 = c * 16;
    int len = (c == NCH - 1) ? 15 : 16;
    // ping-pong buffer selected by offset arithmetic (no pointer arrays!)
    float* Kc = (float*)(smem + ((c & 1) ? OFF_KC1 : OFF_KC0));

#ifdef USE_TDM
    if (len == 16) {
      if (wid == 0) {
        if (c + 1 < NCH - 1) {               // pipeline next full chunk
          tdm_load_chunk(ksb + (long)(c + 1) * 16 * Hdim,
                         (c & 1) ? OFF_KC0 : OFF_KC1);
          __builtin_amdgcn_s_wait_tensorcnt((short)1);   // chunk c complete
        } else {
          __builtin_amdgcn_s_wait_tensorcnt((short)0);
        }
      }
    } else {
      // last (partial) chunk: manual zero-padded load
      for (int i = tid; i < 16 * Hdim / 4; i += 256) {
        int row = i >> 5, c4 = (i & 31) * 4;
        v4f v = {};
        if (row < len) v = *(const v4f*)(ksb + (t0 + row) * Hdim + c4);
        *(v4f*)(Kc + row * 132 + c4) = v;
      }
    }
#else
    for (int i = tid; i < 16 * Hdim / 4; i += 256) {
      int row = i >> 5, c4 = (i & 31) * 4;
      v4f v = {};
      if (row < len) v = *(const v4f*)(ksb + (t0 + row) * Hdim + c4);
      *(v4f*)(Kc + row * 132 + c4) = v;
      if (c + 1 < NCH)
        __builtin_prefetch(ksb + (t0 + 16 + row) * Hdim + c4, 0, 1);
    }
#endif
    __syncthreads();

    // row norms -> Kn (f32), Knbf (A layout), KnTbf (transposed, B layout)
    {
      int row = tid >> 4, j0 = (tid & 15) * 8;
      const float* kp = Kc + row * 132 + j0;
      v4f x0 = *(const v4f*)(kp);
      v4f x1 = *(const v4f*)(kp + 4);
      float s = 0.f;
#pragma unroll
      for (int j = 0; j < 4; ++j) s += x0[j] * x0[j] + x1[j] * x1[j];
      s += __shfl_xor(s, 1); s += __shfl_xor(s, 2);
      s += __shfl_xor(s, 4); s += __shfl_xor(s, 8);
      float inv = 1.0f / fmaxf(sqrtf(s), 1e-12f);
      v4f n0 = x0 * inv, n1 = x1 * inv;
      *(v4f*)(Kn + row * 132 + j0) = n0;
      *(v4f*)(Kn + row * 132 + j0 + 4) = n1;
#pragma unroll
      for (int j = 0; j < 4; ++j) {
        Knbf[row * 136 + j0 + j]       = (__bf16)n0[j];
        Knbf[row * 136 + j0 + 4 + j]   = (__bf16)n1[j];
        KnTbf[(j0 + j) * 24 + row]     = (__bf16)n0[j];
        KnTbf[(j0 + 4 + j) * 24 + row] = (__bf16)n1[j];
      }
    }
    __syncthreads();

    // W = K - Kn @ M^T   (one 16-col tile per wave; A-frags hoisted bf16)
    {
      int ntb = wid * 16;
      v8f cc = load_c_f32(Kc, 132, ntb, lid);
      v16bf a0 = load_a_bf16v(Knbf, 136, 0, lid);
      v16bf a1 = load_a_bf16v(Knbf, 136, 32, lid);
      v16bf a2 = load_a_bf16v(Knbf, 136, 64, lid);
      v16bf a3 = load_a_bf16v(Knbf, 136, 96, lid);
#pragma unroll
      for (int e = 0; e < 16; ++e) {        // negate A = -Kn
        a0[e] = (__bf16)(-(float)a0[e]); a1[e] = (__bf16)(-(float)a1[e]);
        a2[e] = (__bf16)(-(float)a2[e]); a3[e] = (__bf16)(-(float)a3[e]);
      }
      cc = wmma_bf(a0, load_bT_f32v(M, 132, 0,  ntb, lid), cc);
      cc = wmma_bf(a1, load_bT_f32v(M, 132, 32, ntb, lid), cc);
      cc = wmma_bf(a2, load_bT_f32v(M, 132, 64, ntb, lid), cc);
      cc = wmma_bf(a3, load_bT_f32v(M, 132, 96, ntb, lid), cc);
      store_c_f32(Wb, 132, ntb, lid, cc);
    }

    // A = tril(Kn Kn^T, -1) on wave 0
    if (wid == 0) {
      v8f cc = {};
#pragma unroll
      for (int ks = 0; ks < 4; ++ks) {
        v16bf a  = load_a_bf16v(Knbf, 136, ks * 32, lid);
        v16bf bm = load_bT_f32v(Kn, 132, ks * 32, 0, lid);
        cc = wmma_bf(a, bm, cc);
      }
      int scol = lid & 15, rb = (lid < 16) ? 0 : 8;
#pragma unroll
      for (int r = 0; r < 8; ++r) {
        int trow = rb + r;
        Am[trow * 16 + scol] = (scol < trow) ? cc[r] : 0.f;
      }
    }
    __syncthreads();

    // forward substitution per column j; u[] in registers; write UTbf
    if (tid < Hdim) {
      int j = tid;
      float u[16];
#pragma unroll
      for (int t = 0; t < 16; ++t) {
        float acc = Wb[t * 132 + j];
#pragma unroll
        for (int s = 0; s < t; ++s) acc -= Am[t * 16 + s] * u[s];
        u[t] = acc;
        UTbf[j * 24 + t] = (__bf16)acc;
      }
    }
    __syncthreads();

    // M += U^T @ Kn: wave wid owns column tile wid; B-frag loaded ONCE.
    {
      int ntb = wid * 16;
      v16bf bm;
      if (lid < 16) {
        const __bf16* pb = KnTbf + (ntb + lid) * 24;  // B[t][j] = KnT[j][t]
        v8bf b0 = *(const v8bf*)(pb);
        v8bf b1 = *(const v8bf*)(pb + 8);
        bm = __builtin_shufflevector(b0, b1, 0, 1, 2, 3, 4, 5, 6, 7,
                                     8, 9, 10, 11, 12, 13, 14, 15);
      } else {
        v16bf z = {};
        bm = z;                                        // lanes hold K=16..31
      }
      v8bf zz = {};
#pragma unroll
      for (int i = 0; i < 8; ++i) {
        int mtb = i * 16;
        int row = lid & 15;
        const __bf16* pa = UTbf + (mtb + row) * 24;  // A[i][t] = U[t][i]
        v8bf lo = (lid < 16) ? *(const v8bf*)(pa) : zz;
        v8bf hi = (lid < 16) ? *(const v8bf*)(pa + 8) : zz;
        v16bf a = __builtin_shufflevector(lo, hi, 0, 1, 2, 3, 4, 5, 6, 7,
                                          8, 9, 10, 11, 12, 13, 14, 15);
        v8f cc = load_c_f32(M + mtb * 132, 132, ntb, lid);
        cc = wmma_bf(a, bm, cc);
        store_c_f32(M + mtb * 132, 132, ntb, lid, cc);
      }
    }
    __syncthreads();
  }

  // read = M @ q,  q = ks[:, L-1, :]
  {
    float* q = (float*)(smem + OFF_KC0);
    if (tid < Hdim) q[tid] = ksb[(long)(Ldim - 1) * Hdim + tid];
    __syncthreads();
    if (tid < Hdim) {
      const float* mp = M + tid * 132;
      float acc = 0.f;
#pragma unroll 8
      for (int j4 = 0; j4 < Hdim / 4; ++j4) {
        v4f m = *(const v4f*)(mp + j4 * 4);
        acc += m[0] * q[j4 * 4] + m[1] * q[j4 * 4 + 1] +
               m[2] * q[j4 * 4 + 2] + m[3] * q[j4 * 4 + 3];
      }
      readv[b * Hdim + tid] = acc;
    }
  }
}

// ---------------------------------------------------------------------------
// Kernel 3: r2 = read @ Wrp + brp  (tiny, VALU), store bf16 for head GEMM
// ---------------------------------------------------------------------------
__global__ __launch_bounds__(256) void k_head1(const float* __restrict__ readv,
                                               const float* __restrict__ Wrp,
                                               const float* __restrict__ brp,
                                               __bf16* __restrict__ r2bf) {
  int idx = blockIdx.x * 256 + threadIdx.x;
  if (idx < Bdim * Hdim) {
    int b = idx >> 7, j = idx & 127;
    const float* rv = readv + b * Hdim;
    float acc = brp[j];
#pragma unroll 8
    for (int i = 0; i < Hdim; ++i) acc += rv[i] * Wrp[i * Hdim + j];
    r2bf[idx] = (__bf16)acc;
  }
}

// ---------------------------------------------------------------------------
// Kernel 4: out = r2 @ Wout + bout  (32 x 32000, K=128) with WMMA.
// ---------------------------------------------------------------------------
__global__ __launch_bounds__(128) void k_head2(const __bf16* __restrict__ r2bf,
                                               const __bf16* __restrict__ Woutbf,
                                               const float* __restrict__ bout,
                                               float* __restrict__ out) {
  __shared__ __align__(16) __bf16 r2s[32 * 136];
  __shared__ __align__(32) __bf16 Bt[4][4 * 32 * 16];
  int tid = threadIdx.x, wid = tid >> 5, lid = tid & 31;

  for (int i = tid; i < Bdim * Hdim; i += 128)
    r2s[(i >> 7) * 136 + (i & 127)] = r2bf[i];
  __syncthreads();

  int ncb = blockIdx.x * 64 + wid * 16;
  __bf16* bt = Bt[wid];
  for (int ii = lid; ii < 128 * 16; ii += 32) {
    int k = ii >> 4, n = ii & 15;
    int lane = n + (((k >> 4) & 1) << 4);
    bt[((k >> 5) * 32 + lane) * 16 + (k & 15)] = Woutbf[(long)k * Vdim + ncb + n];
  }

  v16bf bfr[4];
#pragma unroll
  for (int ks = 0; ks < 4; ++ks) bfr[ks] = load_b_frag(bt, ks, 1, 0, lid);

  int n = lid & 15;
  float bb = bout[ncb + n];
#pragma unroll
  for (int mt = 0; mt < 2; ++mt) {
    v8f c = {};
#pragma unroll
    for (int ks = 0; ks < 4; ++ks)
      c = wmma_bf(load_a_bf16v(r2s + mt * 16 * 136, 136, ks * 32, lid), bfr[ks], c);
    int rb = (lid < 16) ? 0 : 8;
#pragma unroll
    for (int r = 0; r < 8; ++r)
      out[(long)(mt * 16 + rb + r) * Vdim + ncb + n] = c[r] + bb;
  }
}

// ---------------------------------------------------------------------------
// Host launcher
// ---------------------------------------------------------------------------
extern "C" void kernel_launch(void* const* d_in, const int* in_sizes, int n_in,
                              void* d_out, int out_size, void* d_ws, size_t ws_size,
                              hipStream_t stream) {
  const int*   seq   = (const int*)d_in[0];
  const float* embed = (const float*)d_in[1];
  const float* W1    = (const float*)d_in[2];
  const float* b1    = (const float*)d_in[3];
  const float* W2    = (const float*)d_in[4];
  const float* b2    = (const float*)d_in[5];
  const float* ln_g  = (const float*)d_in[6];
  const float* ln_b  = (const float*)d_in[7];
  const float* Wkp   = (const float*)d_in[8];
  const float* Wrp   = (const float*)d_in[9];
  const float* brp   = (const float*)d_in[10];
  const float* Wout  = (const float*)d_in[11];
  const float* bout  = (const float*)d_in[12];
  float* out = (float*)d_out;

  char* ws = (char*)d_ws;
  size_t off = 0;
  float* ksall = (float*)(ws + off);    off += (size_t)Bdim * Ldim * Hdim * sizeof(float);
  __bf16* Woutbf = (__bf16*)(ws + off); off += (size_t)Hdim * Vdim * sizeof(__bf16);
  float* readv = (float*)(ws + off);    off += (size_t)Bdim * Hdim * sizeof(float);
  __bf16* r2bf = (__bf16*)(ws + off);   off += (size_t)Bdim * Hdim * sizeof(__bf16);

  k_cvt<<<(Hdim * Vdim + 255) / 256, 256, 0, stream>>>(Wout, Woutbf, Hdim * Vdim);

  size_t lds2 = (size_t)(Hdim * TWO_H + TWO_H * Hdim + Hdim * Hdim) * 2
              + (size_t)(TWO_H + 3 * Hdim) * 4
              + (size_t)4 * 16 * 132 * 4
              + (size_t)4 * 16 * 264 * 2
              + (size_t)4 * 16 * 136 * 2;
  k_tokens<<<(Bdim * Ldim) / 64, 128, lds2, stream>>>(seq, embed, W1, b1, W2, b2,
                                                      ln_g, ln_b, Wkp, ksall);

  k_scan<<<Bdim, 256, SCAN_LDS, stream>>>(ksall, readv);

  k_head1<<<(Bdim * Hdim + 255) / 256, 256, 0, stream>>>(readv, Wrp, brp, r2bf);

  k_head2<<<Vdim / 64, 128, 0, stream>>>(r2bf, Woutbf, bout, out);
}
